// AdvectionNCA_6640019440049
// MI455X (gfx1250) — matrix operand order repaired and checked by reading, not compiled, not run on hardware
//
#include <hip/hip_runtime.h>
#include <hip/hip_bf16.h>

typedef __attribute__((ext_vector_type(16))) _Float16 v16h;
typedef __attribute__((ext_vector_type(8)))  float    v8f;
typedef __attribute__((ext_vector_type(4)))  float    v4f;

#define CC      16
#define HH      512
#define WW      512
#define BB      8
#define NPIX    (BB*HH*WW)          // 2,097,152
#define HIDDEN  64
#define H2      32
#define FIRE    0.5f
#define ADV_DT  0.25f
#define DIFF    0.05f
#define VNOISE  0.2f
#define VDAMP   0.95f

// ---------------- RNG helpers (deterministic per (seed, pixel)) -------------
__device__ __forceinline__ unsigned pcg(unsigned v) {
    v = v * 747796405u + 2891336453u;
    unsigned w = ((v >> ((v >> 28u) + 4u)) ^ v) * 277803737u;
    return (w >> 22u) ^ w;
}
__device__ __forceinline__ float u01(unsigned h) {
    return ((h >> 8) + 1u) * (1.0f / 16777217.0f);   // (0,1]
}

// ---------------------------------------------------------------------------
// Kernel A: perception + WMMA MLP + stochastic update + velocity processing.
// block = 128 threads (4 waves), grid = 2048 blocks, 16 tiles/wave, tile = 16px.
// ---------------------------------------------------------------------------
__global__ __launch_bounds__(128)
void nca_mlp_kernel(const float* __restrict__ S, const int* __restrict__ seedp,
                    const float* __restrict__ w1,  const float* __restrict__ b1,
                    const float* __restrict__ w2,  const float* __restrict__ b2,
                    const float* __restrict__ wr,  const float* __restrict__ br,
                    const float* __restrict__ wv,  const float* __restrict__ bv,
                    const float* __restrict__ wh,  const float* __restrict__ bh,
                    float* __restrict__ out,
                    float* __restrict__ vxp, float* __restrict__ vyp,
                    float* __restrict__ m0p)
{
    __shared__ _Float16 pbuf[4][16][64];   // perception  (16 px x 64 ch)
    __shared__ _Float16 xbuf[4][16][64];   // layer outputs staging
    __shared__ float    hbuf[4][16][16];   // head outputs (16 px x 16 cols)

    const int tid  = threadIdx.x;
    const int wv32 = tid >> 5;       // wave in block
    const int lane = tid & 31;
    const int row  = lane & 15;      // M-row / pixel-in-tile for A-frags
    const int hi   = lane >> 4;      // lane half
    const unsigned sd = (unsigned)seedp[0];
    const v4f* __restrict__ S4 = (const v4f*)S;   // 4 x float4 per pixel

    // ---------------- hoisted weight fragments (B-matrix layout) -----------
    // B 16-bit 32x16: col = lane&15, halfs i -> K = kchunk*32 + hi*16 + i
    const int col = lane & 15;
    v16h B1[4][2], B2[2][2], Bh;
    #pragma unroll
    for (int n = 0; n < 4; ++n)
        #pragma unroll
        for (int kc = 0; kc < 2; ++kc) {
            #pragma unroll
            for (int i = 0; i < 16; ++i) {
                int k = kc * 32 + hi * 16 + i;
                B1[n][kc][i] = (_Float16)w1[k * HIDDEN + (n * 16 + col)];
            }
        }
    #pragma unroll
    for (int n = 0; n < 2; ++n)
        #pragma unroll
        for (int kc = 0; kc < 2; ++kc) {
            #pragma unroll
            for (int i = 0; i < 16; ++i) {
                int k = kc * 32 + hi * 16 + i;
                B2[n][kc][i] = (_Float16)w2[k * H2 + (n * 16 + col)];
            }
        }
    #pragma unroll
    for (int i = 0; i < 16; ++i) {
        int k = hi * 16 + i;
        float wvv;
        if      (col < 3)  wvv = wr[k * 3 + col];
        else if (col < 5)  wvv = wv[k * 2 + (col - 3)];
        else if (col < 13) wvv = wh[k * 8 + (col - 5)];
        else               wvv = 0.0f;
        Bh[i] = (_Float16)wvv;
    }
    float hbias;
    if      (col < 3)  hbias = br[col];
    else if (col < 5)  hbias = bv[col - 3];
    else if (col < 13) hbias = bh[col - 5];
    else               hbias = 0.0f;
    const float bias1[4] = { b1[0*16+col], b1[1*16+col], b1[2*16+col], b1[3*16+col] };
    const float bias2[2] = { b2[0*16+col], b2[1*16+col] };

    // 131072 tiles total == 2048 blocks * 4 waves * 16 iterations (uniform)
    for (int it = 0; it < 16; ++it) {
        const int t   = it * 8192 + blockIdx.x * 4 + wv32;
        const int bb  = t >> 14;            // 16384 tiles per image
        const int rem = t & 16383;
        const int y   = rem >> 5;
        const int xt  = (rem & 31) << 4;

        // ------------- perception into LDS (each lane: 1 px, 8 ch) --------
        {
            const int xl = xt + row;
            const int xm = (xl + WW - 1) & (WW - 1), xp = (xl + 1) & (WW - 1);
            const int ym = (y  + HH - 1) & (HH - 1), yp = (y  + 1) & (HH - 1);
            const long base = (long)bb * HH * WW;
            // 9 stencil pixel indices (float4-granule base = pixel*4)
            const long p00 = (base + (long)ym*WW + xm) * 4;
            const long p01 = (base + (long)ym*WW + xl) * 4;
            const long p02 = (base + (long)ym*WW + xp) * 4;
            const long p10 = (base + (long)y *WW + xm) * 4;
            const long p11 = (base + (long)y *WW + xl) * 4;
            const long p12 = (base + (long)y *WW + xp) * 4;
            const long p20 = (base + (long)yp*WW + xm) * 4;
            const long p21 = (base + (long)yp*WW + xl) * 4;
            const long p22 = (base + (long)yp*WW + xp) * 4;
            #pragma unroll
            for (int g4 = 0; g4 < 2; ++g4) {
                const int c4 = hi * 2 + g4;           // float4 index within pixel
                v4f q00 = S4[p00 + c4], q01 = S4[p01 + c4], q02 = S4[p02 + c4];
                v4f q10 = S4[p10 + c4], q11 = S4[p11 + c4], q12 = S4[p12 + c4];
                v4f q20 = S4[p20 + c4], q21 = S4[p21 + c4], q22 = S4[p22 + c4];
                v4f gx  = (q02 + 2.f*q12 + q22 - q00 - 2.f*q10 - q20) * 0.125f;
                v4f gy  = (q20 + 2.f*q21 + q22 - q00 - 2.f*q01 - q02) * 0.125f;
                v4f lap = (q12 + q10 + q21 + q01 - 4.f*q11) * 0.25f;
                #pragma unroll
                for (int e = 0; e < 4; ++e) {
                    const int ch = hi * 8 + g4 * 4 + e;
                    pbuf[wv32][row][     ch] = (_Float16)q11[e];
                    pbuf[wv32][row][16 + ch] = (_Float16)gx[e];
                    pbuf[wv32][row][32 + ch] = (_Float16)gy[e];
                    pbuf[wv32][row][48 + ch] = (_Float16)lap[e];
                }
            }
            // prefetch next tile's stencil rows (global_prefetch_b8)
            if (it < 15) {
                const int tn   = t + 8192;
                const int bbn  = tn >> 14;
                const int remn = tn & 16383;
                const int yn   = remn >> 5;
                const int xtn  = (remn & 31) << 4;
                const long basen = (long)bbn * HH * WW;
                const int xln = xtn + row;
                const int ymn = (yn + HH - 1) & (HH - 1), ypn = (yn + 1) & (HH - 1);
                __builtin_prefetch(&S[(basen + (long)ymn*WW + xln) * 16], 0, 3);
                __builtin_prefetch(&S[(basen + (long)yn *WW + xln) * 16], 0, 3);
                __builtin_prefetch(&S[(basen + (long)ypn*WW + xln) * 16], 0, 3);
            }
        }
        __syncthreads();

        // ------------- layer 1: (16x64) @ (64x64) -------------------------
        // A 16-bit 16x32 frag: half j -> K = kc*32+hi*16+j
        v16h a0, a1;
        #pragma unroll
        for (int j = 0; j < 8; ++j) {
            a0[j]     = pbuf[wv32][row][      hi*16 + j];
            a0[j + 8] = pbuf[wv32][row][8 +   hi*16 + j];
            a1[j]     = pbuf[wv32][row][32 +  hi*16 + j];
            a1[j + 8] = pbuf[wv32][row][40 +  hi*16 + j];
        }
        #pragma unroll
        for (int n = 0; n < 4; ++n) {
            v8f acc = {};
            acc = __builtin_amdgcn_wmma_f32_16x16x32_f16(false, a0, false, B1[n][0],
                                                         (short)0, acc, false, false);
            acc = __builtin_amdgcn_wmma_f32_16x16x32_f16(false, a1, false, B1[n][1],
                                                         (short)0, acc, false, false);
            #pragma unroll
            for (int i = 0; i < 8; ++i) {
                float v = acc[i] + bias1[n];
                v = v > 0.f ? v : 0.f;
                xbuf[wv32][i + hi*8][n*16 + col] = (_Float16)v;   // D layout
            }
        }
        __syncthreads();

        // ------------- layer 2: (16x64) @ (64x32) -------------------------
        #pragma unroll
        for (int j = 0; j < 8; ++j) {
            a0[j]     = xbuf[wv32][row][      hi*16 + j];
            a0[j + 8] = xbuf[wv32][row][8 +   hi*16 + j];
            a1[j]     = xbuf[wv32][row][32 +  hi*16 + j];
            a1[j + 8] = xbuf[wv32][row][40 +  hi*16 + j];
        }
        __syncthreads();
        #pragma unroll
        for (int n = 0; n < 2; ++n) {
            v8f acc = {};
            acc = __builtin_amdgcn_wmma_f32_16x16x32_f16(false, a0, false, B2[n][0],
                                                         (short)0, acc, false, false);
            acc = __builtin_amdgcn_wmma_f32_16x16x32_f16(false, a1, false, B2[n][1],
                                                         (short)0, acc, false, false);
            #pragma unroll
            for (int i = 0; i < 8; ++i) {
                float v = acc[i] + bias2[n];
                v = v > 0.f ? v : 0.f;
                xbuf[wv32][i + hi*8][n*16 + col] = (_Float16)v;
            }
        }
        __syncthreads();

        // ------------- heads: (16x32) @ (32x16) ---------------------------
        #pragma unroll
        for (int j = 0; j < 8; ++j) {
            a0[j]     = xbuf[wv32][row][      hi*16 + j];
            a0[j + 8] = xbuf[wv32][row][8 +   hi*16 + j];
        }
        {
            v8f acc = {};
            acc = __builtin_amdgcn_wmma_f32_16x16x32_f16(false, a0, false, Bh,
                                                         (short)0, acc, false, false);
            #pragma unroll
            for (int i = 0; i < 8; ++i)
                hbuf[wv32][i + hi*8][col] = acc[i] + hbias;
        }
        __syncthreads();

        // ------------- per-pixel epilogue (lanes 0-15) --------------------
        if (lane < 16) {
            const int p = lane;
            const int x = xt + p;
            const long g = (long)bb * HH * WW + (long)y * WW + x;

            float hd[13];
            #pragma unroll
            for (int i = 0; i < 13; ++i) hd[i] = hbuf[wv32][p][i];

            float ds[16];
            ds[0] = hd[0]; ds[1] = hd[1]; ds[2] = hd[2];
            ds[3] = 0.f;
            ds[4] = hd[3]; ds[5] = hd[4];
            ds[6] = 0.f;   ds[7] = 0.f;
            #pragma unroll
            for (int i = 0; i < 8; ++i) ds[8 + i] = hd[5 + i];

            const unsigned gk = (unsigned)g;
            float fire = (u01(pcg(gk * 4u + 0u + sd * 0x9E3779B9u)) < FIRE) ? 1.f : 0.f;
            float uu1 = u01(pcg(gk * 4u + 1u + sd * 0x85EBCA6Bu));
            float uu2 = u01(pcg(gk * 4u + 2u + sd * 0xC2B2AE35u));
            float rr  = sqrtf(-2.0f * logf(uu1));
            float n1  = rr * cosf(6.28318530718f * uu2);
            float n2  = rr * sinf(6.28318530718f * uu2);

            // vectorized 16-channel state read (4 x global_load_b128)
            float ns[16];
            #pragma unroll
            for (int q = 0; q < 4; ++q) {
                v4f sv = S4[g * 4 + q];
                #pragma unroll
                for (int e = 0; e < 4; ++e) ns[q*4 + e] = sv[e] + ds[q*4 + e] * fire;
            }

            float vx = fminf(fmaxf(ns[4], -1.f), 1.f) * VDAMP;
            float vy = fminf(fmaxf(ns[5], -1.f), 1.f) * VDAMP;
            ns[4] = vx; ns[5] = vy;
            ns[0] = fminf(fmaxf(ns[0], 0.f), 1.f);
            ns[1] = fminf(fmaxf(ns[1], 0.f), 1.f);
            ns[2] = fminf(fmaxf(ns[2], 0.f), 1.f);

            // vectorized 16-channel state write (4 x global_store_b128)
            v4f* __restrict__ out4 = (v4f*)out;
            #pragma unroll
            for (int q = 0; q < 4; ++q) {
                v4f ov = { ns[q*4+0], ns[q*4+1], ns[q*4+2], ns[q*4+3] };
                out4[g * 4 + q] = ov;
            }

            vxp[g] = vx + VNOISE * n1;
            vyp[g] = vy + VNOISE * n2;
            m0p[g] = ns[3];          // mass channel untouched by ds
        }
        __syncthreads();
    }
}

// ---------------------------------------------------------------------------
// Kernel B/C: one semi-Lagrangian advection step (torus bilinear gather)
// ---------------------------------------------------------------------------
__global__ __launch_bounds__(256)
void advect_kernel(const float* __restrict__ src,
                   const float* __restrict__ vxp, const float* __restrict__ vyp,
                   float* __restrict__ dst)
{
    const int i = blockIdx.x * blockDim.x + threadIdx.x;
    if (i >= NPIX) return;
    const int bb  = i >> 18;
    const int rem = i & 262143;
    const int y   = rem >> 9;
    const int x   = rem & 511;
    const float sy = (float)y - vyp[i] * ADV_DT;
    const float sx = (float)x - vxp[i] * ADV_DT;
    const float y0f = floorf(sy), x0f = floorf(sx);
    const float fy = sy - y0f,   fx = sx - x0f;
    const int y0 = ((int)y0f) & 511, y1 = (y0 + 1) & 511;
    const int x0 = ((int)x0f) & 511, x1 = (x0 + 1) & 511;
    const long base = (long)bb * HH * WW;
    const float m00 = src[base + (long)y0 * WW + x0];
    const float m01 = src[base + (long)y0 * WW + x1];
    const float m10 = src[base + (long)y1 * WW + x0];
    const float m11 = src[base + (long)y1 * WW + x1];
    dst[i] = (1.f - fy) * ((1.f - fx) * m00 + fx * m01)
           +        fy  * ((1.f - fx) * m10 + fx * m11);
}

// ---------------------------------------------------------------------------
// Kernel D: diffuse mass and write it into d_out channel MASS (=3)
// ---------------------------------------------------------------------------
__global__ __launch_bounds__(256)
void diffuse_kernel(const float* __restrict__ m, float* __restrict__ out)
{
    const int i = blockIdx.x * blockDim.x + threadIdx.x;
    if (i >= NPIX) return;
    const int bb  = i >> 18;
    const int rem = i & 262143;
    const int y   = rem >> 9;
    const int x   = rem & 511;
    const long base = (long)bb * HH * WW;
    const int xm = (x + 511) & 511, xp = (x + 1) & 511;
    const int ym = (y + 511) & 511, yp = (y + 1) & 511;
    const float avg4 = (m[base + (long)y  * WW + xp] + m[base + (long)y  * WW + xm] +
                        m[base + (long)yp * WW + x ] + m[base + (long)ym * WW + x ]) * 0.25f;
    const float mc = m[i];
    out[((long)i << 4) + 3] = mc + DIFF * (avg4 - mc);
}

// ---------------------------------------------------------------------------
extern "C" void kernel_launch(void* const* d_in, const int* in_sizes, int n_in,
                              void* d_out, int out_size, void* d_ws, size_t ws_size,
                              hipStream_t stream)
{
    const float* S    = (const float*)d_in[0];
    const int*   seed = (const int*)  d_in[1];
    const float* w1   = (const float*)d_in[2];
    const float* b1   = (const float*)d_in[3];
    const float* w2   = (const float*)d_in[4];
    const float* b2   = (const float*)d_in[5];
    const float* wr   = (const float*)d_in[6];
    const float* br   = (const float*)d_in[7];
    const float* wv   = (const float*)d_in[8];
    const float* bv   = (const float*)d_in[9];
    const float* wh   = (const float*)d_in[10];
    const float* bh   = (const float*)d_in[11];
    float* out = (float*)d_out;

    float* ws  = (float*)d_ws;
    float* vxp = ws;
    float* vyp = ws + (size_t)NPIX;
    float* m0  = ws + (size_t)NPIX * 2;
    float* m1  = ws + (size_t)NPIX * 3;
    float* m2  = ws + (size_t)NPIX * 4;

    nca_mlp_kernel<<<2048, 128, 0, stream>>>(S, seed, w1, b1, w2, b2,
                                             wr, br, wv, bv, wh, bh,
                                             out, vxp, vyp, m0);

    const int eg = (NPIX + 255) / 256;
    advect_kernel <<<eg, 256, 0, stream>>>(m0, vxp, vyp, m1);
    advect_kernel <<<eg, 256, 0, stream>>>(m1, vxp, vyp, m2);
    diffuse_kernel<<<eg, 256, 0, stream>>>(m2, out);
}
